// EvoformerBlockCore_29841432773273
// MI455X (gfx1250) — matrix operand order repaired and checked by reading, not compiled, not run on hardware
//
#include <hip/hip_runtime.h>
#include <math.h>

typedef _Float16 f16;
typedef __attribute__((ext_vector_type(16))) _Float16 v16h;
typedef __attribute__((ext_vector_type(8)))  _Float16 v8h;
typedef __attribute__((ext_vector_type(8)))  float    v8f;

constexpr int  NSEQc = 128, NRESc = 256, CMc = 256, CZc = 128;
constexpr int  COPMc = 32, CMULc = 128, CATTc = 32, HPc = 4, TNc = 4;
constexpr long SRc = (long)NSEQc * NRESc;   // 32768 rows of m
constexpr long RRc = (long)NRESc * NRESc;   // 65536 rows of z

__device__ __forceinline__ float sigf(float x) { return 1.0f / (1.0f + __expf(-x)); }

// ---------------- WMMA GEMM:  out = A[M,K] * B[N,K]^T (+bias[n]) ----------------
enum { EPI_F32 = 0, EPI_RELU_F16 = 1, EPI_F16 = 2, EPI_OPM_F16 = 3 };

__device__ __forceinline__
void epi_store(v8f acc, int tileM, int tileN, int l16, int hi,
               const float* __restrict__ bias, void* __restrict__ outp,
               long obase, long ors, long ocs, int mode)
{
    const int col = tileN + l16;
    const float bb = bias ? bias[col] : 0.0f;
#pragma unroll
    for (int r = 0; r < 8; ++r) {
        const int row = tileM + hi * 8 + r;    // C/D layout: VGPR r -> rows r / r+8
        float v = acc[r] + bb;
        switch (mode) {
        case EPI_F32:
            ((float*)outp)[obase + (long)row * ors + (long)col * ocs] = v; break;
        case EPI_RELU_F16:
            ((f16*)outp)[obase + (long)row * ors + (long)col * ocs] = (f16)fmaxf(v, 0.0f); break;
        case EPI_F16:
            ((f16*)outp)[obase + (long)row * ors + (long)col * ocs] = (f16)v; break;
        case EPI_OPM_F16: {     // outer[(i,c),(j,d)] -> [i,j, c*32+d] f16
            const long i = row >> 5, cc = row & 31, j = col >> 5, d = col & 31;
            ((f16*)outp)[((i * NRESc + j) << 10) + (cc << 5) + d] = (f16)v; break;
        }
        }
    }
}

// Small-N fallback: 16x16 per wave, 4 waves / block (32x32 block tile).
__global__ __launch_bounds__(128)
void k_wmma_gemm(const f16* __restrict__ A, long lda, long a1, long a2,
                 const f16* __restrict__ B, long ldb, long b1, long b2,
                 const float* __restrict__ bias,
                 void* __restrict__ outp, long ors, long ocs, long o1, long o2,
                 int bdiv, int M, int N, int K, int mode)
{
    const int batch = blockIdx.z;
    const int bq = batch / bdiv, br = batch % bdiv;
    const f16* Ab = A + (long)bq * a1 + (long)br * a2;
    const f16* Bb = B + (long)bq * b1 + (long)br * b2;
    const long obase = (long)bq * o1 + (long)br * o2;

    const int wave = threadIdx.x >> 5;
    const int lane = threadIdx.x & 31;
    const int tileM = (blockIdx.x * 2 + (wave >> 1)) * 16;
    const int tileN = (blockIdx.y * 2 + (wave & 1)) * 16;
    if (tileM >= M || tileN >= N) return;   // wave-uniform: EXEC stays all-1s for WMMA

    const int l16 = lane & 15;
    const int hi  = lane >> 4;
    // A fragment (16x32 f16): lane half 0 holds K 0-15, half 1 holds K 16-31 (contiguous)
    const f16* arow = Ab + (long)(tileM + l16) * lda + hi * 16;
    // B fragment (32x16 f16): lane half 0 holds K 0-15, half 1 holds K 16-31 (contiguous)
    const f16* brow = Bb + (long)(tileN + l16) * ldb + hi * 16;

    v8f acc = {};
    for (int k0 = 0; k0 < K; k0 += 32) {
        union { v16h v; v8h h[2]; } au;
        au.h[0] = *(const v8h*)(arow + k0);
        au.h[1] = *(const v8h*)(arow + k0 + 8);
        v16h bv = *(const v16h*)(brow + k0);
        if (k0 + 96 < K) {
            __builtin_prefetch(arow + k0 + 96, 0, 3);   // global_prefetch_b8
            __builtin_prefetch(brow + k0 + 96, 0, 3);
        }
        acc = __builtin_amdgcn_wmma_f32_16x16x32_f16(false, au.v, false, bv,
                                                     (short)0, acc, false, false);
    }
    epi_store(acc, tileM, tileN, l16, hi, bias, outp, obase, ors, ocs, mode);
}

// Main path: 32x64 per wave (2 A-frags x 4 B-frags x 8 accumulators),
// 8 waves / block arranged 4x2 -> 128x128 block tile. Requires M%128==0, N%128==0.
__global__ __launch_bounds__(256)
void k_wmma_gemm_big(const f16* __restrict__ A, long lda, long a1, long a2,
                     const f16* __restrict__ B, long ldb, long b1, long b2,
                     const float* __restrict__ bias,
                     void* __restrict__ outp, long ors, long ocs, long o1, long o2,
                     int bdiv, int M, int N, int K, int mode)
{
    const int batch = blockIdx.z;
    const int bq = batch / bdiv, br = batch % bdiv;
    const f16* Ab = A + (long)bq * a1 + (long)br * a2;
    const f16* Bb = B + (long)bq * b1 + (long)br * b2;
    const long obase = (long)bq * o1 + (long)br * o2;

    const int wave = threadIdx.x >> 5;          // 0..7
    const int lane = threadIdx.x & 31;
    const int tileM = (blockIdx.x * 4 + (wave >> 1)) * 32;
    const int tileN = (blockIdx.y * 2 + (wave & 1)) * 64;

    const int l16 = lane & 15;
    const int hi  = lane >> 4;
    const f16* arow0 = Ab + (long)(tileM + l16) * lda + hi * 16;
    const f16* arow1 = arow0 + (long)16 * lda;
    const f16* brow0 = Bb + (long)(tileN + l16) * ldb + hi * 16;
    const f16* brow1 = brow0 + (long)16 * ldb;
    const f16* brow2 = brow0 + (long)32 * ldb;
    const f16* brow3 = brow0 + (long)48 * ldb;

    v8f acc[2][4] = {};
    for (int k0 = 0; k0 < K; k0 += 32) {
        union { v16h v; v8h h[2]; } a0u, a1u;
        a0u.h[0] = *(const v8h*)(arow0 + k0);
        a0u.h[1] = *(const v8h*)(arow0 + k0 + 8);
        a1u.h[0] = *(const v8h*)(arow1 + k0);
        a1u.h[1] = *(const v8h*)(arow1 + k0 + 8);
        v16h b0 = *(const v16h*)(brow0 + k0);
        v16h b1 = *(const v16h*)(brow1 + k0);
        v16h b2 = *(const v16h*)(brow2 + k0);
        v16h b3 = *(const v16h*)(brow3 + k0);
        if (k0 + 96 < K) {
            __builtin_prefetch(arow0 + k0 + 96, 0, 3);
            __builtin_prefetch(arow1 + k0 + 96, 0, 3);
            __builtin_prefetch(brow0 + k0 + 96, 0, 3);
            __builtin_prefetch(brow1 + k0 + 96, 0, 3);
            __builtin_prefetch(brow2 + k0 + 96, 0, 3);
            __builtin_prefetch(brow3 + k0 + 96, 0, 3);
        }
        acc[0][0] = __builtin_amdgcn_wmma_f32_16x16x32_f16(false, a0u.v, false, b0, (short)0, acc[0][0], false, false);
        acc[0][1] = __builtin_amdgcn_wmma_f32_16x16x32_f16(false, a0u.v, false, b1, (short)0, acc[0][1], false, false);
        acc[0][2] = __builtin_amdgcn_wmma_f32_16x16x32_f16(false, a0u.v, false, b2, (short)0, acc[0][2], false, false);
        acc[0][3] = __builtin_amdgcn_wmma_f32_16x16x32_f16(false, a0u.v, false, b3, (short)0, acc[0][3], false, false);
        acc[1][0] = __builtin_amdgcn_wmma_f32_16x16x32_f16(false, a1u.v, false, b0, (short)0, acc[1][0], false, false);
        acc[1][1] = __builtin_amdgcn_wmma_f32_16x16x32_f16(false, a1u.v, false, b1, (short)0, acc[1][1], false, false);
        acc[1][2] = __builtin_amdgcn_wmma_f32_16x16x32_f16(false, a1u.v, false, b2, (short)0, acc[1][2], false, false);
        acc[1][3] = __builtin_amdgcn_wmma_f32_16x16x32_f16(false, a1u.v, false, b3, (short)0, acc[1][3], false, false);
    }
#pragma unroll
    for (int mi = 0; mi < 2; ++mi)
#pragma unroll
        for (int nj = 0; nj < 4; ++nj)
            epi_store(acc[mi][nj], tileM + 16 * mi, tileN + 16 * nj, l16, hi,
                      bias, outp, obase, ors, ocs, mode);
}

// ---------------- LayerNorm (f32 in -> f16 out), one wave per row ----------------
__global__ __launch_bounds__(256)
void k_layernorm_f16(const float* __restrict__ x, const float* __restrict__ g,
                     const float* __restrict__ b, f16* __restrict__ out, int C)
{
    const long row = (long)blockIdx.x * 8 + (threadIdx.x >> 5);
    const int lane = threadIdx.x & 31;
    const float* xr = x + row * C;
    float s = 0.f, s2 = 0.f;
    for (int c = lane; c < C; c += 32) { float v = xr[c]; s += v; s2 += v * v; }
    for (int o = 16; o; o >>= 1) { s += __shfl_xor(s, o, 32); s2 += __shfl_xor(s2, o, 32); }
    const float mu = s / C;
    const float inv = rsqrtf(fmaxf(s2 / C - mu * mu, 0.f) + 1e-5f);
    f16* orow = out + row * C;
    for (int c = lane; c < C; c += 32)
        orow[c] = (f16)((xr[c] - mu) * inv * g[c] + b[c]);
}

// ---------------- attention softmax over k (256), one wave per (i,h,q) row -------
__global__ __launch_bounds__(256)
void k_att_softmax(const float* __restrict__ s, const float* __restrict__ mask,
                   const float* __restrict__ btri, f16* __restrict__ probs)
{
    const long row = (long)blockIdx.x * 8 + (threadIdx.x >> 5);
    const int lane = threadIdx.x & 31;
    const int  q = (int)(row & 255);
    const long ih = row >> 8;
    const int  h = (int)(ih & 3);
    const long i = ih >> 2;
    const float* sr = s + row * 256;
    const float* mrow = mask + i * NRESc;
    float vals[8];
    float mx = -1e30f;
#pragma unroll
    for (int t = 0; t < 8; ++t) {
        const int k = lane + t * 32;
        float v = sr[k] + 1e9f * (mrow[k] - 1.0f) + btri[((long)q * 256 + k) * 4 + h];
        vals[t] = v; mx = fmaxf(mx, v);
    }
    for (int o = 16; o; o >>= 1) mx = fmaxf(mx, __shfl_xor(mx, o, 32));
    float sum = 0.f;
#pragma unroll
    for (int t = 0; t < 8; ++t) { vals[t] = __expf(vals[t] - mx); sum += vals[t]; }
    for (int o = 16; o; o >>= 1) sum += __shfl_xor(sum, o, 32);
    const float inv = 1.0f / sum;
    f16* pr = probs + row * 256;
#pragma unroll
    for (int t = 0; t < 8; ++t) pr[lane + t * 32] = (f16)(vals[t] * inv);
}

// ---------------- elementwise / pack kernels ----------------
__global__ void k_cvt_f16(const float* __restrict__ in, f16* __restrict__ out, float sc, long n)
{ long i = (long)blockIdx.x * 256 + threadIdx.x; if (i < n) out[i] = (f16)(in[i] * sc); }

__global__ void k_residual_mask(float* __restrict__ dst, const float* __restrict__ src,
                                const float* __restrict__ y, const float* __restrict__ mask,
                                int cshift, long n)
{ long i = (long)blockIdx.x * 256 + threadIdx.x; if (i < n) dst[i] = src[i] + y[i] * mask[i >> cshift]; }

__global__ void k_addto(float* __restrict__ dst, const float* __restrict__ y, long n)
{ long i = (long)blockIdx.x * 256 + threadIdx.x; if (i < n) dst[i] += y[i]; }

__global__ void k_pack_opm(const float* __restrict__ a, const float* __restrict__ mask,
                           f16* __restrict__ out, long n)   // out[(i*32+c)*128 + s]
{
    long idx = (long)blockIdx.x * 256 + threadIdx.x; if (idx >= n) return;
    const long m2 = idx >> 7; const int s = (int)(idx & 127);
    out[idx] = (f16)(a[(long)s * (NRESc * COPMc) + m2] * mask[(long)s * NRESc + (m2 >> 5)]);
}

__global__ void k_opm_norm(const float* __restrict__ mask, float* __restrict__ nrm, long n)
{
    long idx = (long)blockIdx.x * 256 + threadIdx.x; if (idx >= n) return;
    const int i = (int)(idx >> 8), j = (int)(idx & 255);
    float s = 0.f;
    for (int t = 0; t < NSEQc; ++t) s += mask[t * NRESc + i] * mask[t * NRESc + j];
    nrm[idx] = s;
}

__global__ void k_opm_final(float* __restrict__ zout, const float* __restrict__ zin,
                            const float* __restrict__ o, const float* __restrict__ nrm, long n)
{ long i = (long)blockIdx.x * 256 + threadIdx.x; if (i < n) zout[i] = zin[i] + o[i] / (1e-3f + nrm[i >> 7]); }

__global__ void k_gate_pack(const float* __restrict__ proj, const float* __restrict__ gate,
                            const float* __restrict__ mask, f16* __restrict__ out,
                            int transposed, long n)     // out[c][i][k]
{
    long idx = (long)blockIdx.x * 256 + threadIdx.x; if (idx >= n) return;
    const int c = (int)(idx >> 16); const long rem = idx & 65535;
    const int i = (int)(rem >> 8), k = (int)(rem & 255);
    const long p = transposed ? ((long)k * NRESc + i) : ((long)i * NRESc + k);
    const long src = p * CMULc + c;
    out[idx] = (f16)(mask[p] * sigf(gate[src]) * proj[src]);
}

__global__ void k_gate_residual(float* __restrict__ z, const float* __restrict__ x,
                                const float* __restrict__ g, long n)
{ long i = (long)blockIdx.x * 256 + threadIdx.x; if (i < n) z[i] += x[i] * sigf(g[i]); }

__global__ void k_gate_mul_f16(const float* __restrict__ o, const float* __restrict__ g,
                               f16* __restrict__ out, long n)
{ long i = (long)blockIdx.x * 256 + threadIdx.x; if (i < n) out[i] = (f16)(o[i] * sigf(g[i])); }

__global__ void k_tri_bias(const f16* __restrict__ x, const float* __restrict__ wb,
                           float* __restrict__ out, long n)  // out[(jq*256+jk)*4+h]
{
    long idx = (long)blockIdx.x * 256 + threadIdx.x; if (idx >= n) return;
    const long row = idx >> 2; const int h = (int)(idx & 3);
    const f16* xr = x + row * CZc; const float* w = wb + h * CZc;
    float acc = 0.f;
    for (int c = 0; c < CZc; ++c) acc += (float)xr[c] * w[c];
    out[idx] = acc;
}

__global__ void k_pack_vT(const float* __restrict__ v, f16* __restrict__ out, long n)
{   // out[i][h][c][j] = v[(i*256+j)*128 + h*32+c]
    long idx = (long)blockIdx.x * 256 + threadIdx.x; if (idx >= n) return;
    const int j = (int)(idx & 255); long r = idx >> 8;
    const int c = (int)(r & 31); r >>= 5;
    const int h = (int)(r & 3);  const long i = r >> 2;
    out[idx] = (f16)v[(i * NRESc + j) * CZc + h * CATTc + c];
}

__global__ void k_transpose_zc(const float* __restrict__ in, float* __restrict__ out, long n)
{
    long idx = (long)blockIdx.x * 256 + threadIdx.x; if (idx >= n) return;
    const int c = (int)(idx & 127); long r = idx >> 7;
    const int j = (int)(r & 255); const long i = r >> 8;
    out[((long)j * NRESc + i) * CZc + c] = in[idx];
}

__global__ void k_transpose_add(float* __restrict__ z, const float* __restrict__ d, long n)
{
    long idx = (long)blockIdx.x * 256 + threadIdx.x; if (idx >= n) return;
    const int c = (int)(idx & 127); long r = idx >> 7;
    const int j = (int)(r & 255); const long i = r >> 8;
    z[idx] += d[((long)j * NRESc + i) * CZc + c];
}

__global__ void k_transpose_mask(const float* __restrict__ in, float* __restrict__ out, long n)
{ long idx = (long)blockIdx.x * 256 + threadIdx.x; if (idx < n) out[idx] = in[(idx & 255) * NRESc + (idx >> 8)]; }

// ---------------- host orchestration ----------------
static inline unsigned gblk(long n) { return (unsigned)((n + 255) / 256); }

enum {
    IN_M = 0, IN_Z, IN_MSA_MASK, IN_PAIR_MASK,
    P_MT_LN_G, P_MT_LN_B, P_MT_W1, P_MT_B1, P_MT_W2, P_MT_B2,
    P_OPM_LN_G, P_OPM_LN_B, P_OPM_WA, P_OPM_BA, P_OPM_WB, P_OPM_BB, P_OPM_WO, P_OPM_BO,
    P_TMO = 18, P_TMI = P_TMO + 16, P_TAS = P_TMI + 16, P_TAE = P_TAS + 10,
    P_PT_LN_G = P_TAE + 10, P_PT_LN_B, P_PT_W1, P_PT_B1, P_PT_W2, P_PT_B2
};

constexpr size_t MBy = 1ull << 20;
constexpr size_t OFF_W16 = 0;            // f16 weight arena (8 MB)
constexpr size_t OFF_LN16 = 8 * MBy;     // LN output f16 (24 MB)
constexpr size_t OFF_BIG16 = 32 * MBy;   // hidden / outer / probs f16 (144 MB)
constexpr size_t OFF_PA16 = 176 * MBy;   // a16 / q16 / o16 (24 MB)
constexpr size_t OFF_PB16 = 200 * MBy;   // b16 / k16 (24 MB)
constexpr size_t OFF_PC16 = 224 * MBy;   // vT16 (24 MB)
constexpr size_t OFF_FA = 248 * MBy;     // scores / transition y (272 MB)
constexpr size_t OFF_FB = 520 * MBy;
constexpr size_t OFF_FC = 554 * MBy;
constexpr size_t OFF_FD = 588 * MBy;
constexpr size_t OFF_FE = 622 * MBy;
constexpr size_t OFF_FF = 656 * MBy;
constexpr size_t OFF_FG = 690 * MBy;
constexpr size_t OFF_FH = 724 * MBy;
constexpr size_t OFF_FI = 758 * MBy;     // z-transpose
constexpr size_t OFF_FJ = 792 * MBy;     // tri-bias / norm / maskT

extern "C" void kernel_launch(void* const* d_in, const int* in_sizes, int n_in,
                              void* d_out, int out_size, void* d_ws, size_t ws_size,
                              hipStream_t stream)
{
    (void)in_sizes; (void)n_in; (void)out_size; (void)ws_size;
    char* ws = (char*)d_ws;
    auto F16 = [&](size_t off) { return (f16*)(ws + off); };
    auto F32 = [&](size_t off) { return (float*)(ws + off); };
    auto FIN = [&](int i) { return (const float*)d_in[i]; };

    float* m_out = (float*)d_out;
    float* z_out = m_out + SRc * CMc;

    // ---- convert all GEMM weights to f16 in the W16 arena ----
    size_t wpos = 0;
    auto walloc = [&](size_t nelem) { size_t r = wpos; wpos += (nelem + 63) & ~(size_t)63; return r; };
    auto cvtw = [&](int pidx, size_t nelem) {
        size_t off = walloc(nelem);
        k_cvt_f16<<<gblk((long)nelem), 256, 0, stream>>>(FIN(pidx), F16(OFF_W16) + off, 1.0f, (long)nelem);
        return F16(OFF_W16) + off;
    };
    f16* w_mt1 = cvtw(P_MT_W1, (size_t)TNc * CMc * CMc);
    f16* w_mt2 = cvtw(P_MT_W2, (size_t)CMc * TNc * CMc);
    f16* w_opa = cvtw(P_OPM_WA, (size_t)COPMc * CMc);
    f16* w_opb = cvtw(P_OPM_WB, (size_t)COPMc * CMc);
    f16* w_opo = cvtw(P_OPM_WO, (size_t)CZc * COPMc * COPMc);
    const int tmbase[2] = { P_TMO, P_TMI };
    const int tmoffs[6] = { 2, 4, 6, 8, 10, 14 };   // wap,wag,wbp,wbg,wg,wz
    f16* w_tm[2][6];
    for (int t = 0; t < 2; ++t)
        for (int j = 0; j < 6; ++j) w_tm[t][j] = cvtw(tmbase[t] + tmoffs[j], (size_t)CZc * CZc);
    const int tabase[2] = { P_TAS, P_TAE };
    const int taoffs[5] = { 2, 3, 4, 6, 8 };        // wq,wk,wv,wg,wo
    f16* w_ta[2][5];
    for (int t = 0; t < 2; ++t)
        for (int j = 0; j < 5; ++j) w_ta[t][j] = cvtw(tabase[t] + taoffs[j], (size_t)CZc * CZc);
    f16* w_pt1 = cvtw(P_PT_W1, (size_t)TNc * CZc * CZc);
    f16* w_pt2 = cvtw(P_PT_W2, (size_t)CZc * TNc * CZc);

    auto gemm = [&](const f16* A, long lda, long a1, long a2,
                    const f16* B, long ldb, long b1, long b2,
                    const float* bias, void* out, long ors, long ocs, long o1, long o2,
                    int bdiv, int M, int N, int K, int mode, int batches) {
        if ((M % 128 == 0) && (N % 128 == 0)) {
            dim3 grid((unsigned)(M / 128), (unsigned)(N / 128), (unsigned)batches);
            k_wmma_gemm_big<<<grid, dim3(256), 0, stream>>>(A, lda, a1, a2, B, ldb, b1, b2,
                                                            bias, out, ors, ocs, o1, o2, bdiv, M, N, K, mode);
        } else {
            dim3 grid((unsigned)(M / 32), (unsigned)((N + 31) / 32), (unsigned)batches);
            k_wmma_gemm<<<grid, dim3(128), 0, stream>>>(A, lda, a1, a2, B, ldb, b1, b2,
                                                        bias, out, ors, ocs, o1, o2, bdiv, M, N, K, mode);
        }
    };
    auto ln = [&](const float* x, int gp, int bp, f16* out, long rows, int C) {
        k_layernorm_f16<<<(unsigned)(rows / 8), 256, 0, stream>>>(x, FIN(gp), FIN(bp), out, C);
    };

    // ================= Phase 1: MSA transition =================
    {
        f16* ln16 = F16(OFF_LN16); f16* h16 = F16(OFF_BIG16); float* yb = F32(OFF_FA);
        ln(FIN(IN_M), P_MT_LN_G, P_MT_LN_B, ln16, SRc, CMc);
        gemm(ln16, CMc, 0, 0, w_mt1, CMc, 0, 0, FIN(P_MT_B1), h16, TNc * CMc, 1, 0, 0,
             1, (int)SRc, TNc * CMc, CMc, EPI_RELU_F16, 1);
        gemm(h16, TNc * CMc, 0, 0, w_mt2, TNc * CMc, 0, 0, FIN(P_MT_B2), yb, CMc, 1, 0, 0,
             1, (int)SRc, CMc, TNc * CMc, EPI_F32, 1);
        k_residual_mask<<<gblk(SRc * CMc), 256, 0, stream>>>(m_out, FIN(IN_M), yb, FIN(IN_MSA_MASK), 8, SRc * CMc);
    }

    // ================= Phase 2: outer product mean =================
    {
        f16* ln16 = F16(OFF_LN16);
        float* ab = F32(OFF_FB); float* bb = F32(OFF_FC);
        f16* a16 = F16(OFF_PA16); f16* b16 = F16(OFF_PB16);
        f16* outer16 = F16(OFF_BIG16);
        float* ob = F32(OFF_FD); float* nrm = F32(OFF_FJ);
        ln(m_out, P_OPM_LN_G, P_OPM_LN_B, ln16, SRc, CMc);
        gemm(ln16, CMc, 0, 0, w_opa, CMc, 0, 0, FIN(P_OPM_BA), ab, COPMc, 1, 0, 0, 1, (int)SRc, COPMc, CMc, EPI_F32, 1);
        gemm(ln16, CMc, 0, 0, w_opb, CMc, 0, 0, FIN(P_OPM_BB), bb, COPMc, 1, 0, 0, 1, (int)SRc, COPMc, CMc, EPI_F32, 1);
        const long np = (long)NRESc * COPMc * NSEQc;
        k_pack_opm<<<gblk(np), 256, 0, stream>>>(ab, FIN(IN_MSA_MASK), a16, np);
        k_pack_opm<<<gblk(np), 256, 0, stream>>>(bb, FIN(IN_MSA_MASK), b16, np);
        gemm(a16, NSEQc, 0, 0, b16, NSEQc, 0, 0, nullptr, outer16, 0, 0, 0, 0,
             1, NRESc * COPMc, NRESc * COPMc, NSEQc, EPI_OPM_F16, 1);
        gemm(outer16, COPMc * COPMc, 0, 0, w_opo, COPMc * COPMc, 0, 0, FIN(P_OPM_BO), ob, CZc, 1, 0, 0,
             1, (int)RRc, CZc, COPMc * COPMc, EPI_F32, 1);
        k_opm_norm<<<gblk(RRc), 256, 0, stream>>>(FIN(IN_MSA_MASK), nrm, RRc);
        k_opm_final<<<gblk(RRc * CZc), 256, 0, stream>>>(z_out, FIN(IN_Z), ob, nrm, RRc * CZc);
    }

    // ================= Phase 3/4: triangle multiplication =================
    auto tri_mul = [&](int t) {
        const int pb = tmbase[t];
        f16* zl16 = F16(OFF_LN16);
        float* apf = F32(OFF_FB); float* agf = F32(OFF_FC);
        float* bpf = F32(OFF_FD); float* bgf = F32(OFF_FE);
        float* ggf = F32(OFF_FF); float* xb = F32(OFF_FG);
        f16* a16 = F16(OFF_PA16); f16* b16 = F16(OFF_PB16);
        ln(z_out, pb + 0, pb + 1, zl16, RRc, CZc);
        gemm(zl16, CZc, 0, 0, w_tm[t][0], CZc, 0, 0, FIN(pb + 3), apf, CMULc, 1, 0, 0, 1, (int)RRc, CMULc, CZc, EPI_F32, 1);
        gemm(zl16, CZc, 0, 0, w_tm[t][1], CZc, 0, 0, FIN(pb + 5), agf, CMULc, 1, 0, 0, 1, (int)RRc, CMULc, CZc, EPI_F32, 1);
        gemm(zl16, CZc, 0, 0, w_tm[t][2], CZc, 0, 0, FIN(pb + 7), bpf, CMULc, 1, 0, 0, 1, (int)RRc, CMULc, CZc, EPI_F32, 1);
        gemm(zl16, CZc, 0, 0, w_tm[t][3], CZc, 0, 0, FIN(pb + 9), bgf, CMULc, 1, 0, 0, 1, (int)RRc, CMULc, CZc, EPI_F32, 1);
        gemm(zl16, CZc, 0, 0, w_tm[t][4], CZc, 0, 0, FIN(pb + 11), ggf, CZc, 1, 0, 0, 1, (int)RRc, CZc, CZc, EPI_F32, 1);
        const long n = (long)CMULc * RRc;
        k_gate_pack<<<gblk(n), 256, 0, stream>>>(apf, agf, FIN(IN_PAIR_MASK), a16, t, n);
        k_gate_pack<<<gblk(n), 256, 0, stream>>>(bpf, bgf, FIN(IN_PAIR_MASK), b16, t, n);
        // per-channel batched: x_c = A_c (256x256) * B_c^T, scatter to [i,j,c]
        gemm(a16, NRESc, RRc, 0, b16, NRESc, RRc, 0, nullptr, xb,
             (long)NRESc * CMULc, CMULc, 1, 0, 1, NRESc, NRESc, NRESc, EPI_F32, CMULc);
        f16* x16 = F16(OFF_LN16);
        ln(xb, pb + 12, pb + 13, x16, RRc, CMULc);
        float* xz = F32(OFF_FB);
        gemm(x16, CMULc, 0, 0, w_tm[t][5], CMULc, 0, 0, FIN(pb + 15), xz, CZc, 1, 0, 0, 1, (int)RRc, CZc, CMULc, EPI_F32, 1);
        k_gate_residual<<<gblk(RRc * CZc), 256, 0, stream>>>(z_out, xz, ggf, RRc * CZc);
    };
    tri_mul(0);
    tri_mul(1);

    // ================= Phase 5/6: triangle attention =================
    auto tri_att = [&](int t, const float* zsrc, const float* mask, float* result) {
        const int pb = tabase[t];
        f16* x16 = F16(OFF_LN16);
        float* qf = F32(OFF_FF); float* kf = F32(OFF_FG); float* vf = F32(OFF_FH);
        float* gf = F32(OFF_FE); float* sb = F32(OFF_FA); float* obuf = F32(OFF_FD);
        float* btri = F32(OFF_FJ);
        f16* q16 = F16(OFF_PA16); f16* k16 = F16(OFF_PB16); f16* vT16 = F16(OFF_PC16);
        f16* probs = F16(OFF_BIG16); f16* o16 = F16(OFF_PA16);
        ln(zsrc, pb + 0, pb + 1, x16, RRc, CZc);
        gemm(x16, CZc, 0, 0, w_ta[t][0], CZc, 0, 0, nullptr, qf, CZc, 1, 0, 0, 1, (int)RRc, CZc, CZc, EPI_F32, 1);
        gemm(x16, CZc, 0, 0, w_ta[t][1], CZc, 0, 0, nullptr, kf, CZc, 1, 0, 0, 1, (int)RRc, CZc, CZc, EPI_F32, 1);
        gemm(x16, CZc, 0, 0, w_ta[t][2], CZc, 0, 0, nullptr, vf, CZc, 1, 0, 0, 1, (int)RRc, CZc, CZc, EPI_F32, 1);
        gemm(x16, CZc, 0, 0, w_ta[t][3], CZc, 0, 0, FIN(pb + 7), gf, CZc, 1, 0, 0, 1, (int)RRc, CZc, CZc, EPI_F32, 1);
        k_tri_bias<<<gblk(RRc * HPc), 256, 0, stream>>>(x16, FIN(pb + 5), btri, RRc * HPc);
        k_cvt_f16<<<gblk(RRc * CZc), 256, 0, stream>>>(qf, q16, 0.17677669529663687f, RRc * CZc); // 1/sqrt(32)
        k_cvt_f16<<<gblk(RRc * CZc), 256, 0, stream>>>(kf, k16, 1.0f, RRc * CZc);
        k_pack_vT<<<gblk(RRc * CZc), 256, 0, stream>>>(vf, vT16, RRc * CZc);
        // scores[(i,h)] = q_slice (256x32) * k_slice^T
        gemm(q16, CZc, (long)NRESc * CZc, CATTc, k16, CZc, (long)NRESc * CZc, CATTc, nullptr,
             sb, NRESc, 1, (long)HPc * NRESc * NRESc, (long)NRESc * NRESc,
             HPc, NRESc, NRESc, CATTc, EPI_F32, NRESc * HPc);
        k_att_softmax<<<(unsigned)(NRESc * HPc * NRESc / 8), 256, 0, stream>>>(sb, mask, btri, probs);
        // o[(i,h)] = probs (256x256) * vT_slice^T (32x256)
        gemm(probs, NRESc, (long)HPc * NRESc * NRESc, (long)NRESc * NRESc,
             vT16, NRESc, (long)HPc * CATTc * NRESc, (long)CATTc * NRESc, nullptr,
             obuf, CZc, 1, (long)NRESc * CZc, CATTc,
             HPc, NRESc, CATTc, NRESc, EPI_F32, NRESc * HPc);
        k_gate_mul_f16<<<gblk(RRc * CZc), 256, 0, stream>>>(obuf, gf, o16, RRc * CZc);
        gemm(o16, CZc, 0, 0, w_ta[t][4], CZc, 0, 0, FIN(pb + 9), result, CZc, 1, 0, 0,
             1, (int)RRc, CZc, CZc, EPI_F32, 1);
    };
    {   // starting node
        float* yb = F32(OFF_FB);
        tri_att(0, z_out, FIN(IN_PAIR_MASK), yb);
        k_addto<<<gblk(RRc * CZc), 256, 0, stream>>>(z_out, yb, RRc * CZc);
    }
    {   // ending node: operate on z^T
        float* zt = F32(OFF_FI);
        float* mT = F32(OFF_FJ + 2 * MBy);
        k_transpose_zc<<<gblk(RRc * CZc), 256, 0, stream>>>(z_out, zt, RRc * CZc);
        k_transpose_mask<<<gblk(RRc), 256, 0, stream>>>(FIN(IN_PAIR_MASK), mT, RRc);
        float* yb = F32(OFF_FB);
        tri_att(1, zt, mT, yb);
        k_transpose_add<<<gblk(RRc * CZc), 256, 0, stream>>>(z_out, yb, RRc * CZc);
    }

    // ================= Phase 7: pair transition =================
    {
        f16* x16 = F16(OFF_LN16); f16* h16 = F16(OFF_BIG16); float* yb = F32(OFF_FA);
        ln(z_out, P_PT_LN_G, P_PT_LN_B, x16, RRc, CZc);
        gemm(x16, CZc, 0, 0, w_pt1, CZc, 0, 0, FIN(P_PT_B1), h16, TNc * CZc, 1, 0, 0,
             1, (int)RRc, TNc * CZc, CZc, EPI_RELU_F16, 1);
        gemm(h16, TNc * CZc, 0, 0, w_pt2, TNc * CZc, 0, 0, FIN(P_PT_B2), yb, CZc, 1, 0, 0,
             1, (int)RRc, CZc, TNc * CZc, EPI_F32, 1);
        k_residual_mask<<<gblk(RRc * CZc), 256, 0, stream>>>(z_out, z_out, yb, FIN(IN_PAIR_MASK), 7, RRc * CZc);
    }
}
